// Encoder_60601988546901
// MI455X (gfx1250) — compile-verified, matrix operand order checked by reading, not run on hardware
//
#include <hip/hip_runtime.h>
#include <hip/hip_bf16.h>

// ---------------------------------------------------------------------------
// GCN forward for MI455X (gfx1250, wave32).
//   GEMMs   : v_wmma_f32_16x16x32_bf16, 32x32 output tile per wave (2x2
//             register blocking -> each A/B fragment feeds 2 WMMAs).
//   Scatter : one wave per edge, lanes over features, global_atomic_add_f32.
//   Epilogue: fused self-loop + bias + relu, emits next layer bf16 activations.
// ---------------------------------------------------------------------------

typedef __bf16 bf16_t;
typedef __bf16 v16bf __attribute__((ext_vector_type(16)));
typedef __bf16 v8bf  __attribute__((ext_vector_type(8)));
typedef float  v8f   __attribute__((ext_vector_type(8)));

#define IN_FT   256
#define HID1    400
#define HID2    200
#define OUT_FT  128

// padded K strides (multiples of 32 for WMMA bf16 k-steps)
#define KP0     256   // IN_FT
#define KP1     416   // HID1 padded
#define KP2     224   // HID2 padded
// padded N dims (multiples of 32 for the 2x2 wave tile)
#define NP0     416   // HID1  -> 416
#define NP1     224   // HID2  -> 224
#define NP2     128   // OUT_FT

// ---------------------------------------------------------------------------
__global__ __launch_bounds__(256) void cvt_bf16_kernel(
    const float* __restrict__ in, bf16_t* __restrict__ out, int n) {
  int i = blockIdx.x * 256 + threadIdx.x;
  if (i < n) out[i] = (bf16_t)in[i];
}

// W [K, Nout] fp32 row-major  ->  Wt [Noutp, Kp] bf16, zero padded
__global__ __launch_bounds__(256) void pack_weight_kernel(
    const float* __restrict__ W, bf16_t* __restrict__ Wt,
    int K, int Nout, int Kp, int Noutp) {
  int idx = blockIdx.x * 256 + threadIdx.x;
  if (idx >= Noutp * Kp) return;
  int n = idx / Kp, k = idx % Kp;
  float v = (n < Nout && k < K) ? W[(size_t)k * Nout + n] : 0.0f;
  Wt[idx] = (bf16_t)v;
}

// ---------------------------------------------------------------------------
__global__ __launch_bounds__(256) void deg_kernel(
    const int* __restrict__ dst, const float* __restrict__ ew,
    float* __restrict__ deg, int E) {
  int e = blockIdx.x * 256 + threadIdx.x;
  if (e < E)
    __hip_atomic_fetch_add(&deg[dst[e]], ew[e],
                           __ATOMIC_RELAXED, __HIP_MEMORY_SCOPE_AGENT);
}

__global__ __launch_bounds__(256) void dinv_kernel(
    const float* __restrict__ deg, float* __restrict__ dinv, int n) {
  int i = blockIdx.x * 256 + threadIdx.x;
  if (i < n) dinv[i] = rsqrtf(deg[i] + 1.0f);   // +1 = self-loop weight
}

// ---------------------------------------------------------------------------
// A-fragment (16x32 bf16, ISA layout): lane<16 holds K=kb..kb+7 / kb+16..23,
// lane>=16 the +8 halves; two contiguous 16-byte loads per fragment.
__device__ __forceinline__ v16bf load_frag(const bf16_t* p) {
  v8bf lo = *(const v8bf*)p;
  v8bf hi = *(const v8bf*)(p + 16);
  v16bf f;
#pragma unroll
  for (int i = 0; i < 8; ++i) { f[i] = lo[i]; f[8 + i] = hi[i]; }
  return f;
}

__device__ __forceinline__ v8f wmma_bf16(v16bf a, v16bf b, v8f c) {
  return __builtin_amdgcn_wmma_f32_16x16x32_bf16(
      false, a, false, b, (short)0, c, false, false);
}

// WMMA GEMM: C[M,Nout] = A[M,Kp](bf16) * Bt[Noutp,Kp]^T(bf16)
// Each wave computes a 32x32 output tile (2x2 blocking of 16x16 WMMA tiles).
// MODE 0: fp32 out (ldo = Nout), no bias
// MODE 1: bf16 out (row stride ldo >= Nout, pad pre-zeroed), + bias + relu
template <int MODE>
__global__ __launch_bounds__(256) void wmma_gemm_kernel(
    const bf16_t* __restrict__ A, const bf16_t* __restrict__ Bt,
    const float* __restrict__ bias, void* __restrict__ Cout,
    int M, int Kp, int Nout, int tiles_n, int ldo) {
  const int wid  = blockIdx.x * 8 + (threadIdx.x >> 5);
  const int lane = threadIdx.x & 31;
  const int tm = wid / tiles_n;
  const int tn = wid % tiles_n;
  if (tm * 32 >= M) return;

  const int r  = lane & 15;
  const int hi = lane >> 4;
  const bf16_t* a0 = A  + (size_t)(tm * 32 + r)      * Kp + hi * 8;
  const bf16_t* a1 = A  + (size_t)(tm * 32 + 16 + r) * Kp + hi * 8;
  const bf16_t* b0 = Bt + (size_t)(tn * 32 + r)      * Kp + hi * 8;
  const bf16_t* b1 = Bt + (size_t)(tn * 32 + 16 + r) * Kp + hi * 8;

  v8f c00 = {}, c01 = {}, c10 = {}, c11 = {};
  for (int kb = 0; kb < Kp; kb += 32) {
    v16bf va0 = load_frag(a0 + kb);
    v16bf va1 = load_frag(a1 + kb);
    v16bf vb0 = load_frag(b0 + kb);
    v16bf vb1 = load_frag(b1 + kb);
    c00 = wmma_bf16(va0, vb0, c00);
    c01 = wmma_bf16(va0, vb1, c01);
    c10 = wmma_bf16(va1, vb0, c10);
    c11 = wmma_bf16(va1, vb1, c11);
  }

  // C layout per 16x16 tile: vgpr i -> row = i + 8*hi, col = lane&15
  const v8f cc[2][2] = {{c00, c01}, {c10, c11}};
#pragma unroll
  for (int sn = 0; sn < 2; ++sn) {
    const int n = tn * 32 + sn * 16 + r;
    if (n >= Nout) continue;
    const float bv = (MODE == 1) ? bias[n] : 0.0f;
#pragma unroll
    for (int sm = 0; sm < 2; ++sm) {
      const v8f c = cc[sm][sn];
#pragma unroll
      for (int i = 0; i < 8; ++i) {
        const int m = tm * 32 + sm * 16 + i + hi * 8;   // M multiple of 32
        float v = c[i] + bv;
        if (MODE == 1) {
          v = v > 0.0f ? v : 0.0f;
          ((bf16_t*)Cout)[(size_t)m * ldo + n] = (bf16_t)v;
        } else {
          ((float*)Cout)[(size_t)m * ldo + n] = v;
        }
      }
    }
  }
}

// ---------------------------------------------------------------------------
// one wave per edge: agg[dst] += t[src] * (dinv[src]*ew*dinv[dst])
__global__ __launch_bounds__(256) void scatter_kernel(
    const float* __restrict__ t, const int* __restrict__ src,
    const int* __restrict__ dst, const float* __restrict__ ew,
    const float* __restrict__ dinv, float* __restrict__ agg,
    int E, int F) {
  const int e = blockIdx.x * 8 + (threadIdx.x >> 5);
  if (e >= E) return;
  const int lane = threadIdx.x & 31;
  const int s = src[e];
  const int d = dst[e];
  const float nrm = dinv[s] * ew[e] * dinv[d];
  const float* trow = t + (size_t)s * F;
  float* arow = agg + (size_t)d * F;
  __builtin_prefetch(trow, 0, 1);                 // global_prefetch_b8
  for (int f = lane; f < F; f += 32)
    __hip_atomic_fetch_add(&arow[f], trow[f] * nrm,
                           __ATOMIC_RELAXED, __HIP_MEMORY_SCOPE_AGENT);
}

// ---------------------------------------------------------------------------
// h_next = relu(agg + t*dinv^2 (self-loop) + bias), bf16 out, padded stride
__global__ __launch_bounds__(256) void combine_bf16_kernel(
    const float* __restrict__ agg, const float* __restrict__ t,
    const float* __restrict__ dinv, const float* __restrict__ bias,
    bf16_t* __restrict__ out, int Nn, int F, int ldo) {
  int idx = blockIdx.x * 256 + threadIdx.x;
  if (idx >= Nn * F) return;
  int i = idx / F, f = idx % F;
  float di = dinv[i];
  float v = agg[idx] + t[idx] * di * di + bias[f];
  out[(size_t)i * ldo + f] = (bf16_t)(v > 0.0f ? v : 0.0f);
}

// final: out = relu(agg + t*dinv^2 + bias), fp32
__global__ __launch_bounds__(256) void combine_f32_kernel(
    const float* __restrict__ agg, const float* __restrict__ t,
    const float* __restrict__ dinv, const float* __restrict__ bias,
    float* __restrict__ out, int Nn, int F) {
  int idx = blockIdx.x * 256 + threadIdx.x;
  if (idx >= Nn * F) return;
  int f = idx % F;
  int i = idx / F;
  float di = dinv[i];
  float v = agg[idx] + t[idx] * di * di + bias[f];
  out[idx] = v > 0.0f ? v : 0.0f;
}

// ---------------------------------------------------------------------------
static inline size_t al256(size_t x) { return (x + 255) & ~(size_t)255; }

extern "C" void kernel_launch(void* const* d_in, const int* in_sizes, int n_in,
                              void* d_out, int out_size, void* d_ws, size_t ws_size,
                              hipStream_t stream) {
  const float* x    = (const float*)d_in[0];
  const int*   ei   = (const int*)  d_in[1];   // [2, E]
  const float* ew   = (const float*)d_in[2];
  const float* fc_W = (const float*)d_in[3];
  const float* fc_b = (const float*)d_in[4];
  const float* W1   = (const float*)d_in[5];
  const float* b1   = (const float*)d_in[6];
  const float* W2   = (const float*)d_in[7];
  const float* b2   = (const float*)d_in[8];

  const int Nn = in_sizes[0] / IN_FT;          // 20000 (multiple of 32)
  const int E  = in_sizes[1] / 2;              // 320000
  const int* src = ei;
  const int* dst = ei + E;

  // ---- workspace bump allocator --------------------------------------------
  char* ws = (char*)d_ws;
  size_t off = 0;
  auto alloc = [&](size_t bytes) { void* p = ws + off; off += al256(bytes); return p; };

  float*  deg  = (float*) alloc((size_t)Nn * 4);
  float*  dinv = (float*) alloc((size_t)Nn * 4);
  bf16_t* xb   = (bf16_t*)alloc((size_t)Nn * KP0 * 2);
  bf16_t* fcWt = (bf16_t*)alloc((size_t)NP0 * KP0 * 2);           // 416 x 256
  bf16_t* h0b  = (bf16_t*)alloc((size_t)Nn * KP1 * 2);            // 20000 x 416
  bf16_t* W1t  = (bf16_t*)alloc((size_t)NP1 * KP1 * 2);           // 224 x 416
  float*  t1   = (float*) alloc((size_t)Nn * HID2 * 4);
  float*  agg1 = (float*) alloc((size_t)Nn * HID2 * 4);
  bf16_t* h1b  = (bf16_t*)alloc((size_t)Nn * KP2 * 2);            // 20000 x 224
  bf16_t* W2t  = (bf16_t*)alloc((size_t)NP2 * KP2 * 2);           // 128 x 224
  float*  t2   = (float*) alloc((size_t)Nn * OUT_FT * 4);
  float*  agg2 = (float*) alloc((size_t)Nn * OUT_FT * 4);
  (void)ws_size;

  // ---- zero atomic targets + padded activation buffers ---------------------
  hipMemsetAsync(deg,  0, (size_t)Nn * 4, stream);
  hipMemsetAsync(agg1, 0, (size_t)Nn * HID2 * 4, stream);
  hipMemsetAsync(agg2, 0, (size_t)Nn * OUT_FT * 4, stream);
  hipMemsetAsync(h0b,  0, (size_t)Nn * KP1 * 2, stream);
  hipMemsetAsync(h1b,  0, (size_t)Nn * KP2 * 2, stream);

  // ---- prep: bf16 activations / transposed-packed weights / degrees --------
  {
    int n = Nn * IN_FT;
    cvt_bf16_kernel<<<(n + 255) / 256, 256, 0, stream>>>(x, xb, n);
  }
  pack_weight_kernel<<<((NP0 * KP0) + 255) / 256, 256, 0, stream>>>(
      fc_W, fcWt, IN_FT, HID1, KP0, NP0);
  pack_weight_kernel<<<((NP1 * KP1) + 255) / 256, 256, 0, stream>>>(
      W1, W1t, HID1, HID2, KP1, NP1);
  pack_weight_kernel<<<((NP2 * KP2) + 255) / 256, 256, 0, stream>>>(
      W2, W2t, HID2, OUT_FT, KP2, NP2);

  deg_kernel <<<(E + 255) / 256, 256, 0, stream>>>(dst, ew, deg, E);
  dinv_kernel<<<(Nn + 255) / 256, 256, 0, stream>>>(deg, dinv, Nn);

  // ---- layer 0: h0 = relu(x @ fc_W + fc_b), bf16 padded to 416 -------------
  {
    const int tiles_n = NP0 / 32;                        // 13
    const int waves   = (Nn / 32) * tiles_n;
    wmma_gemm_kernel<1><<<(waves + 7) / 8, 256, 0, stream>>>(
        xb, fcWt, fc_b, h0b, Nn, KP0, HID1, tiles_n, KP1);
  }

  // ---- layer 1: t1 = h0 @ W1 ; scatter ; relu(agg + self + b1) -> h1b ------
  {
    const int tiles_n = NP1 / 32;                        // 7
    const int waves   = (Nn / 32) * tiles_n;
    wmma_gemm_kernel<0><<<(waves + 7) / 8, 256, 0, stream>>>(
        h0b, W1t, nullptr, t1, Nn, KP1, HID2, tiles_n, HID2);
    scatter_kernel<<<(E + 7) / 8, 256, 0, stream>>>(
        t1, src, dst, ew, dinv, agg1, E, HID2);
    int n = Nn * HID2;
    combine_bf16_kernel<<<(n + 255) / 256, 256, 0, stream>>>(
        agg1, t1, dinv, b1, h1b, Nn, HID2, KP2);
  }

  // ---- layer 2: t2 = h1 @ W2 ; scatter ; relu(agg + self + b2) -> out ------
  {
    const int tiles_n = NP2 / 32;                        // 4
    const int waves   = (Nn / 32) * tiles_n;
    wmma_gemm_kernel<0><<<(waves + 7) / 8, 256, 0, stream>>>(
        h1b, W2t, nullptr, t2, Nn, KP2, OUT_FT, tiles_n, OUT_FT);
    scatter_kernel<<<(E + 7) / 8, 256, 0, stream>>>(
        t2, src, dst, ew, dinv, agg2, E, OUT_FT);
    int n = Nn * OUT_FT;
    combine_f32_kernel<<<(n + 255) / 256, 256, 0, stream>>>(
        agg2, t2, dinv, b2, (float*)d_out, Nn, OUT_FT);
  }
}
